// AttentionBlock_19645180412485
// MI455X (gfx1250) — matrix operand order repaired and checked by reading, not compiled, not run on hardware
//
#include <hip/hip_runtime.h>

#define N_TOK  4096
#define C_DIM  128
#define NHEADS 4
#define HDIM   32

typedef __attribute__((ext_vector_type(16))) __bf16 bf16x16;
typedef __attribute__((ext_vector_type(8)))  __bf16 bf16x8;
typedef __attribute__((ext_vector_type(8)))  float  floatx8;

__device__ __forceinline__ floatx8 wmma_bf16(bf16x16 a, bf16x16 b, floatx8 c) {
    return __builtin_amdgcn_wmma_f32_16x16x32_bf16(false, a, false, b, (short)0, c, false, false);
}

// A fragment 16x32 (MxK) from row-major source (lda in elements).
// Layout used here (contiguous, not ISA table order): lane L -> row M=L%16; hf=L/16: elems 0..15 = K(hf*16 .. +15), contiguous
__device__ __forceinline__ bf16x16 load_afrag(const __bf16* base, int lda, int lane) {
    int m = lane & 15, hf = lane >> 4;
    const __bf16* row = base + (size_t)m * lda;
    bf16x8 lo = *reinterpret_cast<const bf16x8*>(row + hf * 16);
    bf16x8 hi = *reinterpret_cast<const bf16x8*>(row + hf * 16 + 8);
    bf16x16 r;
#pragma unroll
    for (int i = 0; i < 8; ++i) { r[i] = lo[i]; r[i + 8] = hi[i]; }
    return r;
}

// B fragment 32x16 (KxN); provider stored row-major as prov[n][k] (ldb in elements).
// Layout used here (same k order as load_afrag): lane L -> column n=L%16; elems 0..15 = K((L/16)*16 .. +15), contiguous in provider.
__device__ __forceinline__ bf16x16 load_bfrag(const __bf16* prov, int ldb, int lane) {
    const __bf16* p = prov + (size_t)(lane & 15) * ldb + (lane >> 4) * 16;
    return *reinterpret_cast<const bf16x16*>(p);
}

// ---------------- GroupNorm -> token-major bf16 h ----------------
__global__ void gn_kernel(const float* __restrict__ x, const float* __restrict__ nw,
                          const float* __restrict__ nb, __bf16* __restrict__ hbf) {
    int b = blockIdx.x >> 5;   // 32 groups per batch
    int g = blockIdx.x & 31;
    const float* base = x + ((size_t)b * C_DIM + g * 4) * N_TOK;
    float s = 0.f, s2 = 0.f;
    for (int i = threadIdx.x; i < 4 * N_TOK; i += 256) {
        float v = base[i];
        s += v; s2 += v * v;
    }
#pragma unroll
    for (int m = 16; m >= 1; m >>= 1) {
        s  += __shfl_xor(s,  m, 32);
        s2 += __shfl_xor(s2, m, 32);
    }
    __shared__ float red[16];
    __shared__ float stats[2];
    int wv = threadIdx.x >> 5;
    if ((threadIdx.x & 31) == 0) { red[wv] = s; red[8 + wv] = s2; }
    __syncthreads();
    if (threadIdx.x == 0) {
        float a = 0.f, b2 = 0.f;
        for (int i = 0; i < 8; ++i) { a += red[i]; b2 += red[8 + i]; }
        float mean = a * (1.f / 16384.f);
        float var  = b2 * (1.f / 16384.f) - mean * mean;
        stats[0] = mean;
        stats[1] = rsqrtf(var + 1e-5f);
    }
    __syncthreads();
    float mean = stats[0], rinv = stats[1];
    for (int i = threadIdx.x; i < 4 * N_TOK; i += 256) {
        int cl = i >> 12;           // channel-in-group
        int n  = i & (N_TOK - 1);
        int c  = g * 4 + cl;
        float v = (base[i] - mean) * rinv * nw[c] + nb[c];
        hbf[((size_t)b * N_TOK + n) * C_DIM + c] = (__bf16)v;
    }
}

__global__ void cvt_kernel(const float* __restrict__ src, __bf16* __restrict__ dst, int n) {
    int i = blockIdx.x * blockDim.x + threadIdx.x;
    if (i < n) dst[i] = (__bf16)src[i];
}

// Interleaved position of token j within its 32-token block:
// pos 2c   <-> token (block + c),      c = 0..15
// pos 2c+1 <-> token (block + 16 + c)
// The attention kernel uses the same interleave for P columns, so P.V is invariant.
__device__ __forceinline__ int vperm(int j) {
    return (j & ~31) | ((j & 15) << 1) | ((j >> 4) & 1);
}

// ---------------- QKV projection: h[8192,128] @ proj_w^T[128,384] ----------------
__global__ void qkv_kernel(const __bf16* __restrict__ hbf, const __bf16* __restrict__ wbf,
                           const float* __restrict__ pb,
                           __bf16* __restrict__ q, __bf16* __restrict__ k,
                           __bf16* __restrict__ vt) {
    int lane = threadIdx.x & 31;
    int gw = blockIdx.x * 8 + (threadIdx.x >> 5);
    int mt = gw / 24, nt = gw - mt * 24;   // 24 = 384/16 column tiles
    int m0 = mt * 16, n0 = nt * 16;
    floatx8 acc = {};
#pragma unroll
    for (int k0 = 0; k0 < C_DIM; k0 += 32) {
        bf16x16 a  = load_afrag(hbf + (size_t)m0 * C_DIM + k0, C_DIM, lane);
        bf16x16 bb = load_bfrag(wbf + (size_t)n0 * C_DIM + k0, C_DIM, lane);
        acc = wmma_bf16(a, bb, acc);
    }
    int nlow = lane & 15;
    int hf   = lane >> 4;
    int head = n0 / 96;                       // wave-uniform (96 % 16 == 0)
    int rb   = n0 - head * 96;                // {0,16,32,48,64,80} wave-uniform
    int sect = __builtin_amdgcn_readfirstlane(rb >> 5);  // 0=Q,1=K,2=V (scalar branch)
    int dd   = (rb & 31) + nlow;              // 0..31 within the section
    float bias = pb[n0 + nlow];
    int bidx = m0 >> 12;                      // tile never crosses batch boundary
    int nt0  = (m0 & (N_TOK - 1)) + 8 * hf;
    size_t bh = (size_t)(bidx * NHEADS + head);
    const float scale = 0.17677669529663687f; // 1/sqrt(32), folded into Q
    if (sect == 0) {
#pragma unroll
        for (int vr = 0; vr < 8; ++vr)
            q[(bh * N_TOK + nt0 + vr) * HDIM + dd] = (__bf16)((acc[vr] + bias) * scale);
    } else if (sect == 1) {
#pragma unroll
        for (int vr = 0; vr < 8; ++vr)
            k[(bh * N_TOK + nt0 + vr) * HDIM + dd] = (__bf16)(acc[vr] + bias);
    } else {
        __bf16* vrow = vt + (bh * HDIM + dd) * N_TOK;   // V^T, interleaved token order
#pragma unroll
        for (int vr = 0; vr < 8; ++vr)
            vrow[vperm(nt0 + vr)] = (__bf16)(acc[vr] + bias);
    }
}

// ---------------- Flash attention: per wave, 16 queries; 128 keys per softmax update ----------------
__global__ void __launch_bounds__(256) attn_kernel(const __bf16* __restrict__ q,
                                                   const __bf16* __restrict__ kk,
                                                   const __bf16* __restrict__ vt,
                                                   __bf16* __restrict__ ao) {
    __shared__ __bf16 pbuf[8][4][16 * 32];   // per-wave P re-layout tiles (C-frag -> A-frag)
    int lane = threadIdx.x & 31;
    int wave = threadIdx.x >> 5;
    int bh = blockIdx.x >> 5;                // (b,h) pair
    int qb = blockIdx.x & 31;                // 32 query blocks of 128
    int m0 = qb * 128 + wave * 16;
    const __bf16* qbase = q  + (size_t)bh * N_TOK * HDIM;
    const __bf16* kbase = kk + (size_t)bh * N_TOK * HDIM;
    const __bf16* vbase = vt + (size_t)bh * HDIM * N_TOK;

    bf16x16 aq = load_afrag(qbase + (size_t)m0 * HDIM, HDIM, lane);
    floatx8 o0 = {}, o1 = {};
    float mst[8], lsum[8];                   // lsum is LANE-PARTIAL (reduced once at the end)
#pragma unroll
    for (int i = 0; i < 8; ++i) { mst[i] = -1e30f; lsum[i] = 0.f; }
    int nlow = lane & 15;
    int rowb = 8 * (lane >> 4);

    for (int j0 = 0; j0 < N_TOK; j0 += 128) {
        floatx8 s[8];
#pragma unroll
        for (int t = 0; t < 8; ++t) {        // S cols j0+16t .. j0+16t+15
            bf16x16 bk = load_bfrag(kbase + (size_t)(j0 + 16 * t) * HDIM, HDIM, lane);
            floatx8 z = {};
            s[t] = wmma_bf16(aq, bk, z);
        }
#pragma unroll
        for (int r = 0; r < 8; ++r) {
            // row = r + rowb; reductions over columns stay inside each 16-lane half
            float mx = s[0][r];
#pragma unroll
            for (int t = 1; t < 8; ++t) mx = fmaxf(mx, s[t][r]);
            mx = fmaxf(mx, __shfl_xor(mx, 1, 32));
            mx = fmaxf(mx, __shfl_xor(mx, 2, 32));
            mx = fmaxf(mx, __shfl_xor(mx, 4, 32));
            mx = fmaxf(mx, __shfl_xor(mx, 8, 32));
            float mnew  = fmaxf(mst[r], mx);
            float alpha = __expf(mst[r] - mnew);
            float p[8], psum = 0.f;
#pragma unroll
            for (int t = 0; t < 8; ++t) { p[t] = __expf(s[t][r] - mnew); psum += p[t]; }
            lsum[r] = lsum[r] * alpha + psum; // lane-partial sum
            mst[r]  = mnew;
            o0[r] *= alpha;
            o1[r] *= alpha;
            int row = r + rowb;
            // Tile tt covers keys j0+32tt..+31, interleaved: col 2c <-> frag 2tt, col 2c+1 <-> frag 2tt+1
#pragma unroll
            for (int tt = 0; tt < 4; ++tt) {
                union { __bf16 h[2]; unsigned u; } pk;
                pk.h[0] = (__bf16)p[2 * tt];
                pk.h[1] = (__bf16)p[2 * tt + 1];
                *reinterpret_cast<unsigned*>(&pbuf[wave][tt][row * 32 + 2 * nlow]) = pk.u;
            }
        }
        __builtin_amdgcn_wave_barrier();
        asm volatile("s_wait_dscnt 0" ::: "memory");   // same-wave LDS writes visible
        bf16x16 ap0 = load_afrag(&pbuf[wave][0][0], 32, lane);
        bf16x16 ap1 = load_afrag(&pbuf[wave][1][0], 32, lane);
        bf16x16 ap2 = load_afrag(&pbuf[wave][2][0], 32, lane);
        bf16x16 ap3 = load_afrag(&pbuf[wave][3][0], 32, lane);
        __builtin_amdgcn_wave_barrier();               // keep next iter's stores after reads
#pragma unroll
        for (int tt = 0; tt < 4; ++tt) {
            bf16x16 ap = (tt == 0) ? ap0 : (tt == 1) ? ap1 : (tt == 2) ? ap2 : ap3;
            // V^T is stored in the same interleaved order, so rows are contiguous
            bf16x16 bv0 = load_bfrag(vbase + j0 + 32 * tt,                      N_TOK, lane); // d 0-15
            bf16x16 bv1 = load_bfrag(vbase + (size_t)16 * N_TOK + j0 + 32 * tt, N_TOK, lane); // d 16-31
            o0 = wmma_bf16(ap, bv0, o0);
            o1 = wmma_bf16(ap, bv1, o1);
        }
    }
    int b = bh >> 2, h = bh & 3;
#pragma unroll
    for (int r = 0; r < 8; ++r) {
        float t = lsum[r];                   // deferred cross-lane row-sum reduction
        t += __shfl_xor(t, 1, 32);
        t += __shfl_xor(t, 2, 32);
        t += __shfl_xor(t, 4, 32);
        t += __shfl_xor(t, 8, 32);
        float inv = 1.f / t;
        int tok = m0 + r + rowb;
        size_t rowoff = ((size_t)b * N_TOK + tok) * C_DIM + h * HDIM;
        ao[rowoff + nlow]      = (__bf16)(o0[r] * inv);
        ao[rowoff + 16 + nlow] = (__bf16)(o1[r] * inv);
    }
}

// ---------------- Output projection + bias + pre-norm residual ----------------
__global__ void oproj_kernel(const __bf16* __restrict__ ao, const __bf16* __restrict__ wo,
                             const float* __restrict__ ob, const float* __restrict__ x,
                             float* __restrict__ out) {
    int lane = threadIdx.x & 31;
    int gw = blockIdx.x * 8 + (threadIdx.x >> 5);
    int mt = gw >> 3, nt = gw & 7;         // 8 = 128/16 column tiles
    int m0 = mt * 16, n0 = nt * 16;
    floatx8 acc = {};
#pragma unroll
    for (int k0 = 0; k0 < C_DIM; k0 += 32) {
        bf16x16 a  = load_afrag(ao + (size_t)m0 * C_DIM + k0, C_DIM, lane);
        bf16x16 bb = load_bfrag(wo + (size_t)n0 * C_DIM + k0, C_DIM, lane);
        acc = wmma_bf16(a, bb, acc);
    }
    int c = n0 + (lane & 15);
    float bias = ob[c];
    int hf = lane >> 4;
#pragma unroll
    for (int r = 0; r < 8; ++r) {
        int tok  = m0 + r + 8 * hf;
        int bidx = tok >> 12;
        int ntok = tok & (N_TOK - 1);
        float resid = x[((size_t)bidx * C_DIM + c) * N_TOK + ntok];  // x_tok transposed read
        out[(size_t)tok * C_DIM + c] = acc[r] + bias + resid;
    }
}

extern "C" void kernel_launch(void* const* d_in, const int* in_sizes, int n_in,
                              void* d_out, int out_size, void* d_ws, size_t ws_size,
                              hipStream_t stream) {
    (void)in_sizes; (void)n_in; (void)out_size; (void)ws_size;
    const float* x     = (const float*)d_in[0];
    // d_in[1] = t : unused by the reference
    const float* nw    = (const float*)d_in[2];
    const float* nb    = (const float*)d_in[3];
    const float* pw    = (const float*)d_in[4];
    const float* pbias = (const float*)d_in[5];
    const float* ow    = (const float*)d_in[6];
    const float* obias = (const float*)d_in[7];
    float* out = (float*)d_out;

    char* p = (char*)d_ws;
    auto carve = [&](size_t bytes) { char* r = p; p += (bytes + 255) & ~(size_t)255; return r; };
    __bf16* hbf  = (__bf16*)carve(8192ull * 128 * 2);       // normalized tokens
    __bf16* wpbf = (__bf16*)carve(384ull  * 128 * 2);       // proj_w bf16
    __bf16* wobf = (__bf16*)carve(128ull  * 128 * 2);       // out_w bf16
    __bf16* qbf  = (__bf16*)carve(8ull * 4096 * 32 * 2);    // Q [BH,N,32] (pre-scaled)
    __bf16* kbf  = (__bf16*)carve(8ull * 4096 * 32 * 2);    // K [BH,N,32]
    __bf16* vtbf = (__bf16*)carve(8ull * 32 * 4096 * 2);    // V^T [BH,32,N] interleaved
    __bf16* aobf = (__bf16*)carve(8192ull * 128 * 2);       // attention output tokens

    gn_kernel<<<dim3(64), dim3(256), 0, stream>>>(x, nw, nb, hbf);
    cvt_kernel<<<dim3(192), dim3(256), 0, stream>>>(pw, wpbf, 384 * 128);
    cvt_kernel<<<dim3(64),  dim3(256), 0, stream>>>(ow, wobf, 128 * 128);
    qkv_kernel<<<dim3(1536), dim3(256), 0, stream>>>(hbf, wpbf, pbias, qbf, kbf, vtbf);
    attn_kernel<<<dim3(256), dim3(256), 0, stream>>>(qbf, kbf, vtbf, aobf);
    oproj_kernel<<<dim3(512), dim3(256), 0, stream>>>(aobf, wobf, obias, x, out);
}
